// HGNN_90177133346939
// MI455X (gfx1250) — matrix operand order repaired and checked by reading, not compiled, not run on hardware
//
#include <hip/hip_runtime.h>
#include <hip/hip_fp16.h>

// ---------------------------------------------------------------------------
// HGNN (Poincare-ball GNN) for MI455X / gfx1250, wave32 + WMMA.
//
// Fused pipeline (memory-bound: minimize HBM bytes, f16 intermediates):
//   K1 encode : feat[i] = concat(emb0[x0], emb1[x1], emb2[x2])         -> f16
//   K2 pack   : W_in / W_out repacked into WMMA B-fragment lane order  -> f16
//   K3 gemm1  : t0 = feat @ W_in + b_in          (exp0 o log0 == id, fused out)
//   K4 agg1   : t1 = log0(relu(exp0(mean_{17} t0[src0])))   (all in one wave)
//   K5 agg2   : t2 = mean_{17} t1[src1]          (log0 o exp0 == id, fused out)
//   K6 gemm2  : out = exp0(t2 @ W_out + b_out)   -> f32 d_out
//
// Structured edges: dst = repeat(arange(N),16) ++ arange(N), so node i's
// in-neighbors are exactly src[16i..16i+15] plus the self loop (degree 17
// for every node) -> direct per-node gather, no atomics / segment machinery.
// ---------------------------------------------------------------------------

typedef __attribute__((ext_vector_type(16))) _Float16 v16h;
typedef __attribute__((ext_vector_type(8)))  _Float16 v8h;
typedef __attribute__((ext_vector_type(8)))  float    v8f;

#define N_NODES 50000
#define DEG     16
#define D_HID   256
#define D_OUT   128
#define KSTEPS  (D_HID / 32)      // 8 WMMA K-steps over K=256
#define EPS_F   1e-5f

// ---------------------------------------------------------------- K1: encode
__global__ void k_encode(const int* __restrict__ x,
                         const float* __restrict__ e0,
                         const float* __restrict__ e1,
                         const float* __restrict__ e2,
                         _Float16* __restrict__ featH) {
    const int node = blockIdx.x;
    const int col  = threadIdx.x;                 // 0..255
    float v;
    if (col < 96)        v = e0[(size_t)x[node * 3 + 0] * 96 + col];
    else if (col < 192)  v = e1[(size_t)x[node * 3 + 1] * 96 + (col - 96)];
    else                 v = e2[(size_t)x[node * 3 + 2] * 64 + (col - 192)];
    featH[(size_t)node * D_HID + col] = (_Float16)v;
}

// ------------------------------------------------ K2: pack weights for WMMA B
// B fragment (16-bit, 32x16): lane l holds column N = l&15, K = (l>>4)*16 + j,
// j = 0..15 contiguous per lane -> one 32-byte load per fragment in the GEMM.
__global__ void k_pack(const float* __restrict__ Win,
                       const float* __restrict__ Wout,
                       _Float16* __restrict__ p1,
                       _Float16* __restrict__ p2) {
    const int e = blockIdx.x * blockDim.x + threadIdx.x;
    if (e < 16 * KSTEPS * 32) {                       // W_in: 16 N-tiles
        const int tileN = e >> 8, kstep = (e >> 5) & 7, lane = e & 31;
        const int col = tileN * 16 + (lane & 15);
        const int kb  = kstep * 32 + (lane >> 4) * 16;
        for (int j = 0; j < 16; ++j)
            p1[(size_t)e * 16 + j] = (_Float16)Win[(size_t)(kb + j) * D_HID + col];
    } else if (e < 16 * KSTEPS * 32 + 8 * KSTEPS * 32) {  // W_out: 8 N-tiles
        const int e2 = e - 16 * KSTEPS * 32;
        const int tileN = e2 >> 8, kstep = (e2 >> 5) & 7, lane = e2 & 31;
        const int col = tileN * 16 + (lane & 15);
        const int kb  = kstep * 32 + (lane >> 4) * 16;
        for (int j = 0; j < 16; ++j)
            p2[(size_t)e2 * 16 + j] = (_Float16)Wout[(size_t)(kb + j) * D_OUT + col];
    }
}

// ----------------------------------------- K3/K6: WMMA GEMM + bias (+ exp0)
// Block = 128 threads (4 waves). Wave w computes NT consecutive 16x16 C tiles;
// the block covers all NCOLS columns of 16 rows so the optional exp0 epilogue
// can reduce row norms inside the block (shfl_xor halves + ds_add_f32).
template <int NT, int NCOLS, bool EXP0, bool OUT_F32>
__global__ void k_gemm(const _Float16* __restrict__ A,
                       const _Float16* __restrict__ Bp,
                       const float* __restrict__ bias,
                       _Float16* __restrict__ outH,
                       float* __restrict__ outF) {
    __shared__ float rowsum[16];
    __shared__ float rowscale[16];

    const int lane  = threadIdx.x & 31;
    const int wave  = threadIdx.x >> 5;
    const int tileM = blockIdx.x;
    const int arow  = tileM * 16 + (lane & 15);     // A: M = lane%16 (both halves)
    const int koffA = (lane >> 4) * 16;             // A: K halves split by lane/16

    if (EXP0) {
        if (threadIdx.x < 16) rowsum[threadIdx.x] = 0.f;
        __syncthreads();
    }

    v8f acc[NT];
    for (int nt = 0; nt < NT; ++nt) acc[nt] = 0.f;

    for (int ks = 0; ks < KSTEPS; ++ks) {
        const _Float16* ap = A + (size_t)arow * D_HID + ks * 32 + koffA;
        v8h lo = *(const v8h*)ap;                   // K = kbase + 0..7
        v8h hi = *(const v8h*)(ap + 8);             // K = kbase + 8..15
        v16h a = __builtin_shufflevector(lo, hi, 0,1,2,3,4,5,6,7,
                                                 8,9,10,11,12,13,14,15);
        for (int nt = 0; nt < NT; ++nt) {
            const int tileN = wave * NT + nt;
            const _Float16* bp =
                Bp + ((size_t)(tileN * KSTEPS + ks) * 32 + lane) * 16;
            v16h b = *(const v16h*)bp;
            acc[nt] = __builtin_amdgcn_wmma_f32_16x16x32_f16(
                false, a, false, b, (short)0, acc[nt], false, false);
        }
    }

    // bias (C/D layout: VGPR r <-> M = r + 8*(lane>=16), N = lane&15)
    const int rb = (lane >> 4) * 8;
    float ps[8];
    for (int r = 0; r < 8; ++r) ps[r] = 0.f;
    for (int nt = 0; nt < NT; ++nt) {
        const int col = (wave * NT + nt) * 16 + (lane & 15);
        const float bv = bias[col];
        for (int r = 0; r < 8; ++r) {
            float v = acc[nt][r] + bv;
            acc[nt][r] = v;
            ps[r] += v * v;
        }
    }

    if (EXP0) {
        // reduce over the 16 lanes sharing a row half (xor stays within halves)
        for (int m = 1; m < 16; m <<= 1)
            for (int r = 0; r < 8; ++r) ps[r] += __shfl_xor(ps[r], m, 32);
        if ((lane & 15) == 0)
            for (int r = 0; r < 8; ++r) atomicAdd(&rowsum[rb + r], ps[r]);
        __syncthreads();
        if (threadIdx.x < 16) {
            float n = sqrtf(rowsum[threadIdx.x]);
            n = fmaxf(n, EPS_F);
            rowscale[threadIdx.x] = tanhf(n) / n;       // exp0 scale
        }
        __syncthreads();
    }

    for (int nt = 0; nt < NT; ++nt) {
        const int col = (wave * NT + nt) * 16 + (lane & 15);
        for (int r = 0; r < 8; ++r) {
            const int rg = tileM * 16 + rb + r;
            const float sc = EXP0 ? rowscale[rb + r] : 1.f;
            const float v  = acc[nt][r] * sc;
            if (OUT_F32) outF[(size_t)rg * NCOLS + col] = v;
            else         outH[(size_t)rg * NCOLS + col] = (_Float16)v;
        }
    }
}

// --------------------------- K4/K5: wave-per-node mean-aggregate over in-edges
// Block = 256 threads = 8 waves; wave handles one node, lane holds 8 columns
// (b128 loads -> 512B contiguous per wave per neighbor row). All norms for the
// fused exp0->relu->log0 (layer 1) reduce with wave32 shfl_xor, no LDS needed.
template <bool LAYER1>
__global__ void k_agg(const _Float16* __restrict__ t,
                      const int* __restrict__ src,
                      _Float16* __restrict__ out) {
    const int wave = threadIdx.x >> 5, lane = threadIdx.x & 31;
    const int i  = blockIdx.x * 8 + wave;            // node (grid*8 == N exactly)
    const int cb = lane * 8;                          // column base for this lane

    float acc[8];
    {
        v8h v = *(const v8h*)(t + (size_t)i * D_HID + cb);   // self loop
        for (int j = 0; j < 8; ++j) acc[j] = (float)v[j];
    }
    for (int k = 0; k < DEG; ++k) {
        const int s = src[i * DEG + k];               // wave-uniform -> s_load
        if (k + 1 < DEG)
            __builtin_prefetch(&t[(size_t)src[i * DEG + k + 1] * D_HID + cb], 0, 1);
        v8h v = *(const v8h*)(t + (size_t)s * D_HID + cb);
        for (int j = 0; j < 8; ++j) acc[j] += (float)v[j];
    }
    for (int j = 0; j < 8; ++j) acc[j] *= (1.f / (float)(DEG + 1));

    if (LAYER1) {
        // exp0: n = max(||v||, EPS), scale = tanh(n)/n
        float s2 = 0.f;
        for (int j = 0; j < 8; ++j) s2 += acc[j] * acc[j];
        for (int m = 1; m < 32; m <<= 1) s2 += __shfl_xor(s2, m, 32);
        float n  = fmaxf(sqrtf(s2), EPS_F);
        float se = tanhf(n) / n;
        // relu in hyperbolic coordinates, accumulating the post-relu norm
        float r2 = 0.f;
        for (int j = 0; j < 8; ++j) {
            acc[j] = fmaxf(acc[j] * se, 0.f);
            r2 += acc[j] * acc[j];
        }
        for (int m = 1; m < 32; m <<= 1) r2 += __shfl_xor(r2, m, 32);
        // log0: m = clip(||r||, EPS, 1-EPS), scale = artanh(m)/m
        float nr = fminf(fmaxf(sqrtf(r2), EPS_F), 1.f - EPS_F);
        float sl = 0.5f * __logf((1.f + nr) / (1.f - nr)) / nr;
        for (int j = 0; j < 8; ++j) acc[j] *= sl;
    }
    // layer-2 path: keep tangent mean directly (log0(exp0(v)) == v)

    v8h o;
    for (int j = 0; j < 8; ++j) o[j] = (_Float16)acc[j];
    *(v8h*)(out + (size_t)i * D_HID + cb) = o;
}

// ---------------------------------------------------------------------------
extern "C" void kernel_launch(void* const* d_in, const int* in_sizes, int n_in,
                              void* d_out, int out_size, void* d_ws, size_t ws_size,
                              hipStream_t stream) {
    const int*   x     = (const int*)  d_in[0];
    const float* emb0  = (const float*)d_in[1];
    const float* emb1  = (const float*)d_in[2];
    const float* emb2  = (const float*)d_in[3];
    const float* W_in  = (const float*)d_in[4];
    const float* b_in  = (const float*)d_in[5];
    const float* W_out = (const float*)d_in[6];
    const float* b_out = (const float*)d_in[7];
    const int*   src0  = (const int*)  d_in[8];
    const int*   src1  = (const int*)  d_in[10];
    float* out = (float*)d_out;

    // workspace carve-out (~77 MB total, 256-B aligned slices)
    char*  ws  = (char*)d_ws;
    size_t off = 0;
    auto alloc = [&](size_t bytes) -> void* {
        void* p = ws + off;
        off += (bytes + 255) & ~(size_t)255;
        return p;
    };
    _Float16* featH = (_Float16*)alloc((size_t)N_NODES * D_HID * 2);
    _Float16* pack1 = (_Float16*)alloc((size_t)16 * KSTEPS * 32 * 16 * 2);
    _Float16* pack2 = (_Float16*)alloc((size_t)8  * KSTEPS * 32 * 16 * 2);
    _Float16* buf0  = (_Float16*)alloc((size_t)N_NODES * D_HID * 2);
    _Float16* buf1  = (_Float16*)alloc((size_t)N_NODES * D_HID * 2);
    _Float16* t2H   = featH;   // feat no longer needed after GEMM1 -> reuse

    k_encode<<<N_NODES, 256, 0, stream>>>(x, emb0, emb1, emb2, featH);
    k_pack<<<(16 * KSTEPS * 32 + 8 * KSTEPS * 32) / 256, 256, 0, stream>>>(
        W_in, W_out, pack1, pack2);

    // t0 = feat @ W_in + b_in   (exp0 of fc_input cancels with layer-1 log0)
    k_gemm<4, D_HID, false, false><<<N_NODES / 16, 128, 0, stream>>>(
        featH, pack1, b_in, buf0, nullptr);

    // t1 = log0(relu(exp0(mean t0[src0])))   — fully fused, wave-per-node
    k_agg<true><<<N_NODES / 8, 256, 0, stream>>>(buf0, src0, buf1);

    // t2 = mean t1[src1]                      — exp0/log0 identity fused out
    k_agg<false><<<N_NODES / 8, 256, 0, stream>>>(buf1, src1, t2H);

    // out = exp0(t2 @ W_out + b_out)          (4 waves x 2 tiles = 128 cols)
    k_gemm<2, D_OUT, true, true><<<N_NODES / 16, 128, 0, stream>>>(
        t2H, pack2, b_out, nullptr, out);
}
